// Trioformer_8899172237444
// MI455X (gfx1250) — matrix operand order repaired and checked by reading, not compiled, not run on hardware
//
#include <hip/hip_runtime.h>

// ---------------------------------------------------------------------------
// Trioformer forward for gfx1250 (MI455X).  All GEMMs / attention matmuls run
// on v_wmma_f32_16x16x32_bf16 (bf16 A/B, f32 accumulate).  Softmax / LN / bias
// math stays f32 on the VALU (v_rcp_f32 / v_rsq_f32 fast paths).  One wave
// owns one 16x16 output tile; register layouts follow CDNA5 ISA 7.12.2
// (16-bit A 16x32, 16-bit B 32x16, f32 C/D).
// ---------------------------------------------------------------------------

typedef __bf16 bf16;
typedef __attribute__((ext_vector_type(16))) __bf16 bf16x16;
typedef __attribute__((ext_vector_type(8)))  __bf16 bf16x8;
typedef __attribute__((ext_vector_type(8)))  float  f32x8;

#define DV 128   // model dim D
#define HC 256   // H*C = HG*CG
#define NH 8     // heads
#define HD 32    // head dim

__device__ __forceinline__ f32x8 wmma_bf(bf16x16 a, bf16x16 b, f32x8 c) {
  return __builtin_amdgcn_wmma_f32_16x16x32_bf16(
      /*neg_a=*/false, a, /*neg_b=*/false, b,
      /*c_mod=*/(short)0, c, /*reuse_a=*/false, /*reuse_b=*/false);
}

__device__ __forceinline__ float fast_rcp(float x)  { return __builtin_amdgcn_rcpf(x); }
__device__ __forceinline__ float fast_rsq(float x)  { return __builtin_amdgcn_rsqf(x); }

// 8 contiguous elements -> av[base..base+7] (16B-aligned by construction)
__device__ __forceinline__ void loadA8(const float* ap, bf16x16& av, int base) {
  float4 f0 = *(const float4*)(ap);
  float4 f1 = *(const float4*)(ap + 4);
  av[base + 0] = (bf16)f0.x; av[base + 1] = (bf16)f0.y;
  av[base + 2] = (bf16)f0.z; av[base + 3] = (bf16)f0.w;
  av[base + 4] = (bf16)f1.x; av[base + 5] = (bf16)f1.y;
  av[base + 6] = (bf16)f1.z; av[base + 7] = (bf16)f1.w;
}
__device__ __forceinline__ void loadA8(const bf16* ap, bf16x16& av, int base) {
  bf16x8 t = *(const bf16x8*)(ap);
#pragma unroll
  for (int j = 0; j < 8; ++j) av[base + j] = t[j];
}

// ---------------------------------------------------------------------------
// Generic GEMM:  D = epi(alpha * A@W [+bias]) [+resid] [*maskRow] -> dst
//   A: [M,K] (TA = float or bf16, row-major, lda); W: [K,N] f32 (ldw)
//   ACT: 0 none / 1 relu / 2 sigmoid.  FLG bits: 1 resid, 2 maskRow,
//   4 accumulate (dst += x), 8 transposed dst-row map over (td0,td1).
// ---------------------------------------------------------------------------
template <typename TA, typename TD, int ACT, int FLG>
__global__ void k_gemm(const TA* __restrict__ A, int lda,
                       const float* __restrict__ W, int ldw,
                       TD* __restrict__ Dst, int ldc,
                       const float* __restrict__ bias,
                       const float* __restrict__ resid,
                       const float* __restrict__ maskRow,
                       int M, int N, int K, float alpha, int td0, int td1) {
  int ntiles = N >> 4;
  int tiles  = (M >> 4) * ntiles;
  int wave = blockIdx.x * (blockDim.x >> 5) + (threadIdx.x >> 5);
  if (wave >= tiles) return;               // whole-wave uniform exit
  int tm = wave / ntiles, tn = wave % ntiles;
  int lane = threadIdx.x & 31, lh = lane >> 4, ll = lane & 15;

  f32x8 acc = {0.f, 0.f, 0.f, 0.f, 0.f, 0.f, 0.f, 0.f};
  int arow = tm * 16 + ll;
  int bcol = tn * 16 + ll;
  for (int k0 = 0; k0 < K; k0 += 32) {
    bf16x16 av, bv;
    const TA* ap = A + (long)arow * lda + k0 + lh * 8;
    loadA8(ap,      av, 0);
    loadA8(ap + 16, av, 8);
    const float* wp = W + (long)(k0 + lh * 8) * ldw + bcol;
#pragma unroll
    for (int j = 0; j < 16; ++j) bv[j] = (bf16)wp[(long)(j < 8 ? j : j + 8) * ldw];
    acc = wmma_bf(av, bv, acc);
  }
  float bn = bias ? bias[bcol] : 0.f;      // lane-invariant across r
#pragma unroll
  for (int r = 0; r < 8; ++r) {
    int m = tm * 16 + lh * 8 + r;
    float x = acc[r] * alpha + bn;
    if (ACT == 1) x = fmaxf(x, 0.f);
    else if (ACT == 2) x = fast_rcp(1.f + __expf(-x));
    if (FLG & 1) x += resid[(long)m * ldc + bcol];
    if (FLG & 2) x *= maskRow[m];
    long drow = (FLG & 8) ? (long)(m % td1) * td0 + (m / td1) : (long)m;
    TD* o = Dst + drow * ldc + bcol;
    if (FLG & 4) *o = (TD)((float)*o + x);
    else         *o = (TD)x;
  }
}

// ---------------------------------------------------------------------------
// LayerNorm over last dim (=128), one wave per row, 4 elems/lane.
// Optional transposed source-row map; optional residual add (dst = addTo+LN).
// ---------------------------------------------------------------------------
template <typename TD>
__global__ void k_ln(const float* __restrict__ src, TD* __restrict__ dst,
                     const float* __restrict__ g, const float* __restrict__ b,
                     const float* __restrict__ addTo,
                     int rows, int trans, int td0, int td1) {
  int wave = blockIdx.x * (blockDim.x >> 5) + (threadIdx.x >> 5);
  if (wave >= rows) return;
  int lane = threadIdx.x & 31;
  long srow = trans ? (long)(wave % td1) * td0 + (wave / td1) : (long)wave;
  const float* x = src + srow * DV + lane * 4;
  float4 vv = *(const float4*)x;
  float v[4] = {vv.x, vv.y, vv.z, vv.w};
  float mu = v[0] + v[1] + v[2] + v[3];
#pragma unroll
  for (int off = 16; off >= 1; off >>= 1) mu += __shfl_xor(mu, off, 32);
  mu *= (1.f / DV);
  float dv[4] = {v[0] - mu, v[1] - mu, v[2] - mu, v[3] - mu};
  float var = dv[0] * dv[0] + dv[1] * dv[1] + dv[2] * dv[2] + dv[3] * dv[3];
#pragma unroll
  for (int off = 16; off >= 1; off >>= 1) var += __shfl_xor(var, off, 32);
  var *= (1.f / DV);
  float rs = fast_rsq(var + 1e-5f);
  long obase = (long)wave * DV + lane * 4;
#pragma unroll
  for (int j = 0; j < 4; ++j) {
    float y = dv[j] * rs * g[lane * 4 + j] + b[lane * 4 + j];
    if (addTo) y += addTo[obase + j];
    dst[obase + j] = (TD)y;
  }
}

// ---------------------------------------------------------------------------
// MHA-pair-bias attention core (per head, per 16-query tile).  q pre-scaled.
// logits = q@k^T + (maskq*maskk*1e9 - 1) + zb[pair,h]; softmax; out = p@v.
// NKT = Lk/16 (10 or 12).  Probs re-layout C->A through per-wave LDS slice
// (DS ops from one wave are in-order; no barrier needed).
// ---------------------------------------------------------------------------
template <int NKT>
__global__ void k_mha_attn(const bf16* __restrict__ q, const bf16* __restrict__ k,
                           const bf16* __restrict__ v,
                           const float* __restrict__ zb, int zbtrans,
                           const float* __restrict__ maskq,
                           const float* __restrict__ maskk,
                           float* __restrict__ out, int Lq) {
  const int Lk = NKT * 16;
  __shared__ bf16 sp[4][16][NKT * 16];
  int wid = threadIdx.x >> 5;
  int qtiles = Lq >> 4;
  int wave = blockIdx.x * 4 + wid;
  if (wave >= NH * qtiles) return;
  int qt = wave % qtiles, h = wave / qtiles;
  int lane = threadIdx.x & 31, lh = lane >> 4, ll = lane & 15;

  bf16x16 aq;
  const bf16* qp = q + (long)(qt * 16 + ll) * HC + h * HD + lh * 8;
  loadA8(qp, aq, 0);
  loadA8(qp + 16, aq, 8);

  f32x8 la[NKT];
#pragma unroll
  for (int t = 0; t < NKT; ++t) {
    bf16x16 bk;
    const bf16* kp = k + (long)(t * 16 + ll) * HC + h * HD + lh * 8;
#pragma unroll
    for (int j = 0; j < 16; ++j) bk[j] = kp[j < 8 ? j : j + 8];
    f32x8 z8 = {0.f, 0.f, 0.f, 0.f, 0.f, 0.f, 0.f, 0.f};
    la[t] = wmma_bf(aq, bk, z8);
  }
#pragma unroll
  for (int r = 0; r < 8; ++r) {
    int m = qt * 16 + lh * 8 + r;
    float mq = maskq[m];
    float rm = -3.4e38f;
#pragma unroll
    for (int t = 0; t < NKT; ++t) {
      int n = t * 16 + ll;
      long zrow = zbtrans ? (long)n * Lq + m : (long)m * Lk + n;
      float x = la[t][r] + (mq * maskk[n] * 1e9f - 1.f) + zb[zrow * 16 + h];
      la[t][r] = x;
      rm = fmaxf(rm, x);
    }
#pragma unroll
    for (int off = 8; off >= 1; off >>= 1) rm = fmaxf(rm, __shfl_xor(rm, off, 32));
    float s = 0.f;
#pragma unroll
    for (int t = 0; t < NKT; ++t) { float e = __expf(la[t][r] - rm); la[t][r] = e; s += e; }
#pragma unroll
    for (int off = 8; off >= 1; off >>= 1) s += __shfl_xor(s, off, 32);
    float is = fast_rcp(s);
#pragma unroll
    for (int t = 0; t < NKT; ++t) sp[wid][lh * 8 + r][t * 16 + ll] = (bf16)(la[t][r] * is);
  }
  // probs @ v  (two 16-col output tiles = 32-wide head)
  f32x8 o0 = {0.f, 0.f, 0.f, 0.f, 0.f, 0.f, 0.f, 0.f};
  f32x8 o1 = {0.f, 0.f, 0.f, 0.f, 0.f, 0.f, 0.f, 0.f};
#pragma unroll
  for (int k0 = 0; k0 < Lk; k0 += 32) {
    bf16x16 ap;
    loadA8(&sp[wid][ll][k0 + lh * 8], ap, 0);
    loadA8(&sp[wid][ll][k0 + lh * 8 + 16], ap, 8);
    bf16x16 bv0, bv1;
#pragma unroll
    for (int j = 0; j < 16; ++j) {
      const bf16* vp = v + (long)(k0 + lh * 8 + (j < 8 ? j : j + 8)) * HC + h * HD;
      bv0[j] = vp[ll];
      bv1[j] = vp[16 + ll];
    }
    o0 = wmma_bf(ap, bv0, o0);
    o1 = wmma_bf(ap, bv1, o1);
  }
#pragma unroll
  for (int r = 0; r < 8; ++r) {
    long orow = (long)(qt * 16 + lh * 8 + r) * HC + h * HD;
    out[orow + ll]      = o0[r];
    out[orow + 16 + ll] = o1[r];
  }
}

// ---------------------------------------------------------------------------
// GAPU attention core.  Batch dim I, keys N = NKT*16.
// logits = q@k^T + znb[i*N+q,h] + d[q,k]*wd[h] + 1e9*(mask[i,k]-1)
// After p@v, gate with wab (holds sigmoid gate), writing g*wa back in place.
// ---------------------------------------------------------------------------
template <int NKT>
__global__ void k_gapu_attn(const bf16* __restrict__ q, const bf16* __restrict__ k,
                            const bf16* __restrict__ v,
                            const float* __restrict__ znb,
                            const float* __restrict__ dmat,
                            const float* __restrict__ wd,
                            const float* __restrict__ maskI,
                            bf16* __restrict__ wab, int I) {
  const int N = NKT * 16;
  __shared__ bf16 sp[4][16][NKT * 16];
  int wid = threadIdx.x >> 5;
  int wave = blockIdx.x * 4 + wid;
  int total = I * NH * NKT;
  if (wave >= total) return;
  int qt = wave % NKT;
  int h  = (wave / NKT) % NH;
  int i  = wave / (NKT * NH);
  long rbase = (long)i * N;
  int lane = threadIdx.x & 31, lh = lane >> 4, ll = lane & 15;
  float wdh = wd[h];

  bf16x16 aq;
  const bf16* qp = q + (rbase + qt * 16 + ll) * HC + h * HD + lh * 8;
  loadA8(qp, aq, 0);
  loadA8(qp + 16, aq, 8);

  f32x8 la[NKT];
#pragma unroll
  for (int t = 0; t < NKT; ++t) {
    bf16x16 bk;
    const bf16* kp = k + (rbase + t * 16 + ll) * HC + h * HD + lh * 8;
#pragma unroll
    for (int j = 0; j < 16; ++j) bk[j] = kp[j < 8 ? j : j + 8];
    f32x8 z8 = {0.f, 0.f, 0.f, 0.f, 0.f, 0.f, 0.f, 0.f};
    la[t] = wmma_bf(aq, bk, z8);
  }
#pragma unroll
  for (int r = 0; r < 8; ++r) {
    int m = qt * 16 + lh * 8 + r;
    float bq = znb[(rbase + m) * 16 + h];
    float rm = -3.4e38f;
#pragma unroll
    for (int t = 0; t < NKT; ++t) {
      int n = t * 16 + ll;
      float x = la[t][r] + bq + dmat[(long)m * N + n] * wdh
              + 1e9f * (maskI[rbase + n] - 1.f);
      la[t][r] = x;
      rm = fmaxf(rm, x);
    }
#pragma unroll
    for (int off = 8; off >= 1; off >>= 1) rm = fmaxf(rm, __shfl_xor(rm, off, 32));
    float s = 0.f;
#pragma unroll
    for (int t = 0; t < NKT; ++t) { float e = __expf(la[t][r] - rm); la[t][r] = e; s += e; }
#pragma unroll
    for (int off = 8; off >= 1; off >>= 1) s += __shfl_xor(s, off, 32);
    float is = fast_rcp(s);
#pragma unroll
    for (int t = 0; t < NKT; ++t) sp[wid][lh * 8 + r][t * 16 + ll] = (bf16)(la[t][r] * is);
  }
  f32x8 o0 = {0.f, 0.f, 0.f, 0.f, 0.f, 0.f, 0.f, 0.f};
  f32x8 o1 = {0.f, 0.f, 0.f, 0.f, 0.f, 0.f, 0.f, 0.f};
#pragma unroll
  for (int k0 = 0; k0 < N; k0 += 32) {
    bf16x16 ap;
    loadA8(&sp[wid][ll][k0 + lh * 8], ap, 0);
    loadA8(&sp[wid][ll][k0 + lh * 8 + 16], ap, 8);
    bf16x16 bv0, bv1;
#pragma unroll
    for (int j = 0; j < 16; ++j) {
      const bf16* vp = v + (rbase + k0 + lh * 8 + (j < 8 ? j : j + 8)) * HC + h * HD;
      bv0[j] = vp[ll];
      bv1[j] = vp[16 + ll];
    }
    o0 = wmma_bf(ap, bv0, o0);
    o1 = wmma_bf(ap, bv1, o1);
  }
#pragma unroll
  for (int r = 0; r < 8; ++r) {
    long orow = (rbase + qt * 16 + lh * 8 + r) * HC + h * HD;
    long i0 = orow + ll, i1 = orow + 16 + ll;
    float g0 = (float)wab[i0], g1 = (float)wab[i1];
    wab[i0] = (bf16)(g0 * o0[r]);
    wab[i1] = (bf16)(g1 * o1[r]);
  }
}

// OPM fused with z update: zpost = zout = z + (a·b/16)*wf + bf
__global__ void k_opm(const float* __restrict__ zin, const float* __restrict__ a,
                      const float* __restrict__ bvec, const float* __restrict__ wf,
                      const float* __restrict__ bf_, float* __restrict__ zpost,
                      float* __restrict__ zout) {
  int lp = blockIdx.x;           // l*160 + p
  int l = lp / 160, p = lp % 160;
  int d = threadIdx.x;
  float m = 0.f;
#pragma unroll
  for (int c = 0; c < 16; ++c) m += a[l * 16 + c] * bvec[p * 16 + c];
  m *= (1.f / 16.f);
  long idx = (long)lp * DV + d;
  float z = zin[idx] + m * wf[d] + bf_[d];
  zpost[idx] = z;
  zout[idx]  = z;
}

__global__ void k_padwb(const float* __restrict__ wb, float* __restrict__ out) {
  int i = blockIdx.x * blockDim.x + threadIdx.x;
  if (i >= 128 * 16) return;
  int r = i >> 4, c = i & 15;
  out[i] = (c < 8) ? wb[r * 8 + c] : 0.f;
}

__global__ void k_maskz(const float* __restrict__ ml, const float* __restrict__ mp,
                        float* __restrict__ mz, float* __restrict__ mzT) {
  int i = blockIdx.x * blockDim.x + threadIdx.x;
  if (i >= 192 * 160) return;
  int l = i / 160, p = i % 160;
  float v = ml[l] * mp[p];
  mz[i] = v;
  mzT[p * 192 + l] = v;
}

static inline int divup(int a, int b) { return (a + b - 1) / b; }

// ---------------------------------------------------------------------------
// Host orchestration.  Input flattening assumed in setup_inputs() insertion
// order, with nested `params` dict leaves also in insertion order:
// 0 h_l  1 h_p  2 z_ij  3 d_ik  4 d_jk_  5 mask_l  6 mask_p
// 7..14  mha_l: wq wk wv wb wo bo ln_g ln_b      15..22 mha_p (same)
// 23,24 trans_l w1 w2   25,26 trans_p w1 w2
// 27..34 opm: lnl_g lnl_b lnp_g lnp_b wl wp wf bf
// 35..45 gapu_l: ln_g ln_b wq wk wv wb wd wg bg wo bo   46..56 gapu_p (same)
// d_out = h_l(192*128) ++ h_p(160*128) ++ z(192*160*128), all f32.
// ---------------------------------------------------------------------------
extern "C" void kernel_launch(void* const* d_in, const int* in_sizes, int n_in,
                              void* d_out, int out_size, void* d_ws, size_t ws_size,
                              hipStream_t stream) {
  (void)in_sizes; (void)out_size; (void)ws_size;
  const int L = 192, P = 160, Z = L * P;
  const float r8 = 0.35355339059327373f;  // 8^-0.5

  const float* in[64];
  for (int i = 0; i < n_in && i < 64; ++i) in[i] = (const float*)d_in[i];
  const float *h_l = in[0], *h_p = in[1], *z_in = in[2];
  const float *d_ik = in[3], *d_jk = in[4], *mask_l = in[5], *mask_p = in[6];

  struct MHAP { const float *wq, *wk, *wv, *wb, *wo, *bo, *lng, *lnb; };
  struct GAPP { const float *lng, *lnb, *wq, *wk, *wv, *wb, *wd, *wg, *bg, *wo, *bo; };
  MHAP ml = {in[7], in[8], in[9], in[10], in[11], in[12], in[13], in[14]};
  MHAP mp = {in[15], in[16], in[17], in[18], in[19], in[20], in[21], in[22]};
  const float *tl_w1 = in[23], *tl_w2 = in[24], *tp_w1 = in[25], *tp_w2 = in[26];
  const float *o_lnlg = in[27], *o_lnlb = in[28], *o_lnpg = in[29], *o_lnpb = in[30];
  const float *o_wl = in[31], *o_wp = in[32], *o_wf = in[33], *o_bf = in[34];
  GAPP gl = {in[35], in[36], in[37], in[38], in[39], in[40], in[41], in[42], in[43], in[44], in[45]};
  GAPP gp = {in[46], in[47], in[48], in[49], in[50], in[51], in[52], in[53], in[54], in[55], in[56]};

  float* out_hl = (float*)d_out;
  float* out_hp = out_hl + L * DV;
  float* out_z  = out_hp + P * DV;

  // scratch bump allocator
  char* wsp = (char*)d_ws;
  size_t off = 0;
  auto alloc = [&](size_t bytes) -> void* {
    void* p = wsp + off;
    off += (bytes + 255) & ~(size_t)255;
    return p;
  };
  float* wbpad_ml = (float*)alloc(2048 * 4);
  float* wbpad_mp = (float*)alloc(2048 * 4);
  float* wbpad_gl = (float*)alloc(2048 * 4);
  float* wbpad_gp = (float*)alloc(2048 * 4);
  float* zb_l  = (float*)alloc((size_t)Z * 16 * 4);
  float* zb_p  = (float*)alloc((size_t)Z * 16 * 4);
  float* znb   = (float*)alloc((size_t)Z * 16 * 4);
  float* attno_l = (float*)alloc((size_t)L * HC * 4);
  float* attno_p = (float*)alloc((size_t)P * HC * 4);
  float* tmp_l = (float*)alloc((size_t)L * DV * 4);
  float* tmp_p = (float*)alloc((size_t)P * DV * 4);
  float* hl1   = (float*)alloc((size_t)L * DV * 4);
  float* hp1   = (float*)alloc((size_t)P * DV * 4);
  float* a_    = (float*)alloc((size_t)L * 16 * 4);
  float* b_    = (float*)alloc((size_t)P * 16 * 4);
  float* mz    = (float*)alloc((size_t)Z * 4);
  float* mzT   = (float*)alloc((size_t)Z * 4);
  float* zpost = (float*)alloc((size_t)Z * DV * 4);
  bf16* q_l = (bf16*)alloc((size_t)L * HC * 2);
  bf16* k_l = (bf16*)alloc((size_t)P * HC * 2);
  bf16* v_l = (bf16*)alloc((size_t)P * HC * 2);
  bf16* q_p = (bf16*)alloc((size_t)P * HC * 2);
  bf16* k_p = (bf16*)alloc((size_t)L * HC * 2);
  bf16* v_p = (bf16*)alloc((size_t)L * HC * 2);
  bf16* t_l = (bf16*)alloc((size_t)L * HC * 2);
  bf16* t_p = (bf16*)alloc((size_t)P * HC * 2);
  bf16* lnl = (bf16*)alloc((size_t)L * DV * 2);
  bf16* lnp = (bf16*)alloc((size_t)P * DV * 2);
  bf16* zn  = (bf16*)alloc((size_t)Z * DV * 2);
  bf16* qg  = (bf16*)alloc((size_t)Z * HC * 2);
  bf16* kg  = (bf16*)alloc((size_t)Z * HC * 2);
  bf16* vg  = (bf16*)alloc((size_t)Z * HC * 2);
  bf16* wab = (bf16*)alloc((size_t)Z * HC * 2);

#define GEMM(TA, TD, ACT, FLG, A, LDA, W, LDW, DST, LDC, BIAS, RES, MR, M, N, K, AL, T0, T1) \
  k_gemm<TA, TD, ACT, FLG><<<divup(((M) / 16) * ((N) / 16), 8), 256, 0, stream>>>(           \
      A, LDA, W, LDW, DST, LDC, BIAS, RES, MR, M, N, K, AL, T0, T1)

  // setup: padded wb's, mask outer products
  k_padwb<<<8, 256, 0, stream>>>(ml.wb, wbpad_ml);
  k_padwb<<<8, 256, 0, stream>>>(mp.wb, wbpad_mp);
  k_padwb<<<8, 256, 0, stream>>>(gl.wb, wbpad_gl);
  k_padwb<<<8, 256, 0, stream>>>(gp.wb, wbpad_gp);
  k_maskz<<<divup(Z, 256), 256, 0, stream>>>(mask_l, mask_p, mz, mzT);

  // ---- MHA (both directions computed from original h) ----
  GEMM(float, bf16, 0, 0, h_l, DV, ml.wq, HC, q_l, HC, nullptr, nullptr, nullptr, L, HC, DV, r8, 0, 0);
  GEMM(float, bf16, 0, 0, h_p, DV, ml.wk, HC, k_l, HC, nullptr, nullptr, nullptr, P, HC, DV, 1.f, 0, 0);
  GEMM(float, bf16, 0, 0, h_p, DV, ml.wv, HC, v_l, HC, nullptr, nullptr, nullptr, P, HC, DV, 1.f, 0, 0);
  GEMM(float, bf16, 0, 0, h_p, DV, mp.wq, HC, q_p, HC, nullptr, nullptr, nullptr, P, HC, DV, r8, 0, 0);
  GEMM(float, bf16, 0, 0, h_l, DV, mp.wk, HC, k_p, HC, nullptr, nullptr, nullptr, L, HC, DV, 1.f, 0, 0);
  GEMM(float, bf16, 0, 0, h_l, DV, mp.wv, HC, v_p, HC, nullptr, nullptr, nullptr, L, HC, DV, 1.f, 0, 0);
  GEMM(float, float, 0, 0, z_in, DV, wbpad_ml, 16, zb_l, 16, nullptr, nullptr, nullptr, Z, 16, DV, 1.f, 0, 0);
  GEMM(float, float, 0, 0, z_in, DV, wbpad_mp, 16, zb_p, 16, nullptr, nullptr, nullptr, Z, 16, DV, 1.f, 0, 0);
  k_mha_attn<10><<<divup(NH * (L / 16), 4), 128, 0, stream>>>(q_l, k_l, v_l, zb_l, 0, mask_l, mask_p, attno_l, L);
  k_mha_attn<12><<<divup(NH * (P / 16), 4), 128, 0, stream>>>(q_p, k_p, v_p, zb_p, 1, mask_p, mask_l, attno_p, P);
  GEMM(float, float, 0, 0, attno_l, HC, ml.wo, DV, tmp_l, DV, ml.bo, nullptr, nullptr, L, DV, HC, 1.f, 0, 0);
  GEMM(float, float, 0, 0, attno_p, HC, mp.wo, DV, tmp_p, DV, mp.bo, nullptr, nullptr, P, DV, HC, 1.f, 0, 0);
  k_ln<float><<<divup(L, 8), 256, 0, stream>>>(tmp_l, hl1, ml.lng, ml.lnb, h_l, L, 0, 0, 0);
  k_ln<float><<<divup(P, 8), 256, 0, stream>>>(tmp_p, hp1, mp.lng, mp.lnb, h_p, P, 0, 0, 0);

  // ---- transitions (write final h into d_out) ----
  GEMM(float, bf16, 1, 0, hl1, DV, tl_w1, HC, t_l, HC, nullptr, nullptr, nullptr, L, HC, DV, 1.f, 0, 0);
  GEMM(bf16, float, 0, 1, t_l, HC, tl_w2, DV, out_hl, DV, nullptr, hl1, nullptr, L, DV, HC, 1.f, 0, 0);
  GEMM(float, bf16, 1, 0, hp1, DV, tp_w1, HC, t_p, HC, nullptr, nullptr, nullptr, P, HC, DV, 1.f, 0, 0);
  GEMM(bf16, float, 0, 1, t_p, HC, tp_w2, DV, out_hp, DV, nullptr, hp1, nullptr, P, DV, HC, 1.f, 0, 0);

  // ---- OPM: z_post = z + outer(a,b)/16 * wf + bf ----
  k_ln<bf16><<<divup(L, 8), 256, 0, stream>>>(out_hl, lnl, o_lnlg, o_lnlb, nullptr, L, 0, 0, 0);
  k_ln<bf16><<<divup(P, 8), 256, 0, stream>>>(out_hp, lnp, o_lnpg, o_lnpb, nullptr, P, 0, 0, 0);
  GEMM(bf16, float, 0, 0, lnl, DV, o_wl, 16, a_, 16, nullptr, nullptr, nullptr, L, 16, DV, 1.f, 0, 0);
  GEMM(bf16, float, 0, 0, lnp, DV, o_wp, 16, b_, 16, nullptr, nullptr, nullptr, P, 16, DV, 1.f, 0, 0);
  k_opm<<<Z, 128, 0, stream>>>(z_in, a_, b_, o_wf, o_bf, zpost, out_z);

  // ---- GAPU on z^T (I=P=160, N=L=192), accumulate transposed into out_z ----
  k_ln<bf16><<<divup(Z, 8), 256, 0, stream>>>(zpost, zn, gl.lng, gl.lnb, nullptr, Z, 1, 160, 192);
  GEMM(bf16, bf16, 0, 0, zn, DV, gl.wq, HC, qg, HC, nullptr, nullptr, nullptr, Z, HC, DV, r8, 0, 0);
  GEMM(bf16, bf16, 0, 0, zn, DV, gl.wk, HC, kg, HC, nullptr, nullptr, nullptr, Z, HC, DV, 1.f, 0, 0);
  GEMM(bf16, bf16, 0, 0, zn, DV, gl.wv, HC, vg, HC, nullptr, nullptr, nullptr, Z, HC, DV, 1.f, 0, 0);
  GEMM(bf16, bf16, 2, 0, zn, DV, gl.wg, HC, wab, HC, gl.bg, nullptr, nullptr, Z, HC, DV, 1.f, 0, 0);
  GEMM(bf16, float, 0, 0, zn, DV, wbpad_gl, 16, znb, 16, nullptr, nullptr, nullptr, Z, 16, DV, 1.f, 0, 0);
  k_gapu_attn<12><<<divup(160 * NH * 12, 4), 128, 0, stream>>>(qg, kg, vg, znb, d_ik, gl.wd, mzT, wab, 160);
  GEMM(bf16, float, 0, 14, wab, HC, gl.wo, DV, out_z, DV, gl.bo, nullptr, mzT, Z, DV, HC, 1.f, 160, 192);

  // ---- GAPU on z (I=L=192, N=P=160), accumulate into out_z ----
  k_ln<bf16><<<divup(Z, 8), 256, 0, stream>>>(zpost, zn, gp.lng, gp.lnb, nullptr, Z, 0, 0, 0);
  GEMM(bf16, bf16, 0, 0, zn, DV, gp.wq, HC, qg, HC, nullptr, nullptr, nullptr, Z, HC, DV, r8, 0, 0);
  GEMM(bf16, bf16, 0, 0, zn, DV, gp.wk, HC, kg, HC, nullptr, nullptr, nullptr, Z, HC, DV, 1.f, 0, 0);
  GEMM(bf16, bf16, 0, 0, zn, DV, gp.wv, HC, vg, HC, nullptr, nullptr, nullptr, Z, HC, DV, 1.f, 0, 0);
  GEMM(bf16, bf16, 2, 0, zn, DV, gp.wg, HC, wab, HC, gp.bg, nullptr, nullptr, Z, HC, DV, 1.f, 0, 0);
  GEMM(bf16, float, 0, 0, zn, DV, wbpad_gp, 16, znb, 16, nullptr, nullptr, nullptr, Z, 16, DV, 1.f, 0, 0);
  k_gapu_attn<10><<<divup(192 * NH * 10, 4), 128, 0, stream>>>(qg, kg, vg, znb, d_jk, gp.wd, mz, wab, 192);
  GEMM(bf16, float, 0, 6, wab, HC, gp.wo, DV, out_z, DV, gp.bo, nullptr, mz, Z, DV, HC, 1.f, 0, 0);
#undef GEMM
}
